// CausalCosmosBlock_78623671321202
// MI455X (gfx1250) — matrix operand order repaired and checked by reading, not compiled, not run on hardware
//
#include <hip/hip_runtime.h>
#include <hip/hip_bf16.h>

// ---- problem constants (from reference setup_inputs) ----
#define BD   2      // batch
#define LQ   256    // new-chunk length
#define DM   4096   // model dim
#define NH   32     // heads
#define DH   128    // head dim
#define SK   128    // sink
#define SC   4224   // cache length S = sink + local
#define SML  3968   // S - L
#define BL   512    // B * L

typedef __attribute__((ext_vector_type(16))) __bf16 v16bf;
typedef __attribute__((ext_vector_type(8)))  float  v8f;

#define WMMA_BF16(a, b, c) \
  __builtin_amdgcn_wmma_f32_16x16x32_bf16(false, (a), false, (b), (short)0, (c), false, false)

// rounded fp32->bf16 (epilogue-quality)
__device__ __forceinline__ unsigned short f2bf(float f) {
  union { float f; unsigned u; } x; x.f = f;
  unsigned r = x.u + 0x7FFFu + ((x.u >> 16) & 1u);
  return (unsigned short)(r >> 16);
}
// truncating fp32->bf16 (cheap, for GEMM operand staging)
__device__ __forceinline__ unsigned short f2bft(float f) {
  union { float f; unsigned u; } x; x.f = f;
  return (unsigned short)(x.u >> 16);
}
__device__ __forceinline__ unsigned pack2t(float a, float b) {
  union { float f; unsigned u; } x, y; x.f = a; y.f = b;
  return (x.u >> 16) | (y.u & 0xFFFF0000u);
}

union Frag { v16bf v; uint4 u[2]; };

// =====================================================================
// Kernel 1: fused QKV GEMM (bf16 WMMA) + per-head RMSNorm epilogue.
// Register double-buffering: global loads of tile t+1 fly during WMMAs
// of tile t. Block: 256 thr (8 waves); tile 128x128 (= one head).
// grid.x: 96 (32 heads x {q,k,v}), grid.y: 4 (M tiles over B*L=512)
// =====================================================================
__global__ __launch_bounds__(256)
void qkv_kernel(const float* __restrict__ x,  const float* __restrict__ Wq,
                const float* __restrict__ Wk, const float* __restrict__ Wv,
                const float* __restrict__ qw, const float* __restrict__ kw,
                unsigned short* __restrict__ qb, unsigned short* __restrict__ kb,
                unsigned short* __restrict__ vb)
{
  const int nt    = blockIdx.x;
  const int mt    = blockIdx.y;
  const int which = nt >> 5;        // 0=q 1=k 2=v
  const int head  = nt & 31;
  const int m0    = mt * 128;
  const int n0    = head * 128;
  const float* W  = (which == 0) ? Wq : ((which == 1) ? Wk : Wv);

  __shared__ unsigned short As[128][40];  // [row][k]  bf16, padded
  __shared__ unsigned short Bs[128][40];  // [n][k]    bf16 (transposed), padded

  const int tid  = threadIdx.x;
  const int lane = tid & 31;
  const int wv   = tid >> 5;
  const int hh   = lane >> 4;
  const int ll   = lane & 15;

  // staging coords
  const int a_c4 = (tid & 7) * 4;      // A: 4 contiguous k
  const int a_rb = tid >> 3;           // A: row base (0..31), 4 passes
  const int b_c4 = (tid & 31) * 4;     // B: 4 contiguous n
  const int b_kr = tid >> 5;           // B: k base (0..7), 4 passes

  v8f acc[8] = {};

  float4 rA[4], rB[4];
  // prologue: prefetch tile 0
  for (int p = 0; p < 4; ++p) {
    rA[p] = *reinterpret_cast<const float4*>(&x[(size_t)(m0 + a_rb + 32 * p) * DM + a_c4]);
    rB[p] = *reinterpret_cast<const float4*>(&W[(size_t)(b_kr + 8 * p) * DM + n0 + b_c4]);
  }

  for (int k0 = 0; k0 < DM; k0 += 32) {
    __syncthreads();   // LDS free (previous compute done)
    // store prefetched tile to LDS (fp32 -> bf16, B transposed)
    for (int p = 0; p < 4; ++p) {
      *reinterpret_cast<uint2*>(&As[a_rb + 32 * p][a_c4]) =
          make_uint2(pack2t(rA[p].x, rA[p].y), pack2t(rA[p].z, rA[p].w));
      const int k = b_kr + 8 * p;
      Bs[b_c4 + 0][k] = f2bft(rB[p].x); Bs[b_c4 + 1][k] = f2bft(rB[p].y);
      Bs[b_c4 + 2][k] = f2bft(rB[p].z); Bs[b_c4 + 3][k] = f2bft(rB[p].w);
    }
    __syncthreads();   // staging visible
    // prefetch tile t+1 (overlaps WMMAs below)
    if (k0 + 32 < DM) {
      const int kn = k0 + 32;
      for (int p = 0; p < 4; ++p) {
        rA[p] = *reinterpret_cast<const float4*>(&x[(size_t)(m0 + a_rb + 32 * p) * DM + kn + a_c4]);
        rB[p] = *reinterpret_cast<const float4*>(&W[(size_t)(kn + b_kr + 8 * p) * DM + n0 + b_c4]);
      }
    }

    Frag af;
    {
      const int row  = wv * 16 + ll;
      const int koff = hh * 8;
      af.u[0] = *reinterpret_cast<const uint4*>(&As[row][koff]);
      af.u[1] = *reinterpret_cast<const uint4*>(&As[row][koff + 16]);
    }
    for (int nc = 0; nc < 8; ++nc) {
      Frag bf_;
      const int n   = nc * 16 + ll;
      const int kb2 = hh * 8;
      bf_.u[0] = *reinterpret_cast<const uint4*>(&Bs[n][kb2]);
      bf_.u[1] = *reinterpret_cast<const uint4*>(&Bs[n][kb2 + 16]);
      acc[nc] = WMMA_BF16(af.v, bf_.v, acc[nc]);
    }
  }

  // epilogue: C layout -> lane holds 8 rows (r + 8*hh), col = 16*nc + ll
  const int rowg = m0 + wv * 16 + hh * 8;
  if (which == 2) {
    for (int r = 0; r < 8; ++r) {
      const int m = rowg + r, b = m >> 8, l = m & 255;
      const size_t base = (((size_t)b * NH + head) * LQ + l) * DH;
      for (int nc = 0; nc < 8; ++nc)
        vb[base + nc * 16 + ll] = f2bf(acc[nc][r]);
    }
  } else {
    const float* nw = (which == 0) ? qw : kw;
    unsigned short* dst = (which == 0) ? qb : kb;
    for (int r = 0; r < 8; ++r) {
      float ss = 0.f;
      for (int nc = 0; nc < 8; ++nc) ss += acc[nc][r] * acc[nc][r];
      ss += __shfl_xor(ss, 1, 32);  // xor<16 stays within the 16-lane half
      ss += __shfl_xor(ss, 2, 32);
      ss += __shfl_xor(ss, 4, 32);
      ss += __shfl_xor(ss, 8, 32);
      const float sc = rsqrtf(ss * (1.0f / 128.0f) + 1e-6f);
      const int m = rowg + r, b = m >> 8, l = m & 255;
      const size_t base = (((size_t)b * NH + head) * LQ + l) * DH;
      for (int nc = 0; nc < 8; ++nc) {
        const int dh = nc * 16 + ll;
        dst[base + dh] = f2bf(acc[nc][r] * sc * nw[dh]);
      }
    }
  }
}

// =====================================================================
// Kernel 2: flash attention over rolled cache (S=4224), bf16 WMMA.
// Block: 256 thr (8 waves) owns (b, h, 128 queries); KV stream is
// L2-resident (138MB < 192MB L2), staged fp32->bf16 per 32-key tile.
// grid.x = B*H*2 = 128
// =====================================================================
__global__ __launch_bounds__(256)
void attn_kernel(const float* __restrict__ cache_k, const float* __restrict__ cache_v,
                 const unsigned short* __restrict__ qb, const unsigned short* __restrict__ kb,
                 const unsigned short* __restrict__ vb, unsigned short* __restrict__ attn)
{
  const int bid = blockIdx.x;
  const int b   = bid >> 6;
  const int h   = (bid >> 1) & 31;
  const int qt  = bid & 1;
  const int q0  = qt * 128;

  __shared__ unsigned short Ks[32][136];     // [key][dh]   bf16
  __shared__ unsigned short Vt[128][40];     // [dh][key]   bf16 (transposed)
  __shared__ unsigned short Pl[8][16][40];   // per-wave P tile 16x32

  const int tid = threadIdx.x;
  const int lane = tid & 31, wv = tid >> 5;
  const int hh = lane >> 4, ll = lane & 15;

  // preload q A-fragments (4 Dh-chunks of 32)
  Frag qf[4];
  {
    const int l = q0 + wv * 16 + ll;
    const size_t base = (((size_t)b * NH + h) * LQ + l) * DH;
    const int koff = hh * 8;
    for (int c = 0; c < 4; ++c) {
      qf[c].u[0] = *reinterpret_cast<const uint4*>(&qb[base + c * 32 + koff]);
      qf[c].u[1] = *reinterpret_cast<const uint4*>(&qb[base + c * 32 + koff + 16]);
    }
  }

  float mrow[8], lrow[8];
  for (int r = 0; r < 8; ++r) { mrow[r] = -3.0e38f; lrow[r] = 0.f; }
  v8f oacc[8] = {};

  const float scale = 0.08838834764831845f;   // 1/sqrt(128)
  const int li0 = q0 + wv * 16 + hh * 8;
  const int jmax_blk = SML + q0 + 127;

  for (int j0 = 0; j0 < SC; j0 += 32) {
    if (j0 > jmax_blk) break;                 // block-uniform early exit
    __syncthreads();
    { // stage K (row-major) and V (transposed) tiles with cache roll remap
      const int key = tid >> 3;
      const int c4  = (tid & 7) * 4;
      const int j   = j0 + key;
      for (int p = 0; p < 4; ++p) {
        const int dh = c4 + 32 * p;
        if (j < SML) {
          const int js = (j < SK) ? j : (j + LQ);
          const size_t g = (((size_t)b * SC + js) * NH + h) * DH + dh;
          const float4 kf = *reinterpret_cast<const float4*>(&cache_k[g]);
          const float4 vf = *reinterpret_cast<const float4*>(&cache_v[g]);
          *reinterpret_cast<uint2*>(&Ks[key][dh]) =
              make_uint2(pack2t(kf.x, kf.y), pack2t(kf.z, kf.w));
          Vt[dh + 0][key] = f2bft(vf.x); Vt[dh + 1][key] = f2bft(vf.y);
          Vt[dh + 2][key] = f2bft(vf.z); Vt[dh + 3][key] = f2bft(vf.w);
        } else {
          const size_t g = (((size_t)b * NH + h) * LQ + (j - SML)) * DH + dh;
          const uint2 kk = *reinterpret_cast<const uint2*>(&kb[g]);
          const uint2 vv = *reinterpret_cast<const uint2*>(&vb[g]);
          *reinterpret_cast<uint2*>(&Ks[key][dh]) = kk;
          Vt[dh + 0][key] = (unsigned short)(vv.x & 0xFFFFu);
          Vt[dh + 1][key] = (unsigned short)(vv.x >> 16);
          Vt[dh + 2][key] = (unsigned short)(vv.y & 0xFFFFu);
          Vt[dh + 3][key] = (unsigned short)(vv.y >> 16);
        }
      }
    }
    __syncthreads();

    // logits: q (16xDh) @ K^T -> two 16x16 tiles (keys 0-15, 16-31)
    v8f c0 = {}, c1 = {};
    for (int c = 0; c < 4; ++c) {
      const int kb2 = c * 32 + hh * 8;
      Frag bk0, bk1;
      bk0.u[0] = *reinterpret_cast<const uint4*>(&Ks[ll][kb2]);
      bk0.u[1] = *reinterpret_cast<const uint4*>(&Ks[ll][kb2 + 16]);
      c0 = WMMA_BF16(qf[c].v, bk0.v, c0);
      bk1.u[0] = *reinterpret_cast<const uint4*>(&Ks[16 + ll][kb2]);
      bk1.u[1] = *reinterpret_cast<const uint4*>(&Ks[16 + ll][kb2 + 16]);
      c1 = WMMA_BF16(qf[c].v, bk1.v, c1);
    }

    // online softmax (cross-lane reduce within each 16-lane half)
    float e0[8], e1[8], al[8];
    for (int r = 0; r < 8; ++r) {
      const int jlim = SML + li0 + r;
      float v0 = c0[r] * scale;
      float v1 = c1[r] * scale;
      if (j0 + ll > jlim)      v0 = -3.0e38f;
      if (j0 + 16 + ll > jlim) v1 = -3.0e38f;
      float pm = fmaxf(v0, v1);
      pm = fmaxf(pm, __shfl_xor(pm, 1, 32));
      pm = fmaxf(pm, __shfl_xor(pm, 2, 32));
      pm = fmaxf(pm, __shfl_xor(pm, 4, 32));
      pm = fmaxf(pm, __shfl_xor(pm, 8, 32));
      const float mnew = fmaxf(mrow[r], pm);
      e0[r] = __expf(v0 - mnew);
      e1[r] = __expf(v1 - mnew);
      float ps = e0[r] + e1[r];
      ps += __shfl_xor(ps, 1, 32);
      ps += __shfl_xor(ps, 2, 32);
      ps += __shfl_xor(ps, 4, 32);
      ps += __shfl_xor(ps, 8, 32);
      al[r]   = __expf(mrow[r] - mnew);
      lrow[r] = lrow[r] * al[r] + ps;
      mrow[r] = mnew;
    }
    for (int nc = 0; nc < 8; ++nc)
      for (int r = 0; r < 8; ++r) oacc[nc][r] *= al[r];

    // C-layout -> A-fragment layout via per-wave LDS round trip
    for (int r = 0; r < 8; ++r) {
      Pl[wv][hh * 8 + r][ll]      = f2bft(e0[r]);
      Pl[wv][hh * 8 + r][16 + ll] = f2bft(e1[r]);
    }
    __syncthreads();
    Frag pf;
    pf.u[0] = *reinterpret_cast<const uint4*>(&Pl[wv][ll][hh * 8]);
    pf.u[1] = *reinterpret_cast<const uint4*>(&Pl[wv][ll][hh * 8 + 16]);

    // O += P (16x32) @ V (32xDh)
    for (int nc = 0; nc < 8; ++nc) {
      Frag bv;
      bv.u[0] = *reinterpret_cast<const uint4*>(&Vt[nc * 16 + ll][hh * 8]);
      bv.u[1] = *reinterpret_cast<const uint4*>(&Vt[nc * 16 + ll][hh * 8 + 16]);
      oacc[nc] = WMMA_BF16(pf.v, bv.v, oacc[nc]);
    }
  }

  // normalize and store bf16 attention output [B*L, H*Dh]
  for (int r = 0; r < 8; ++r) {
    const int li = li0 + r;
    const float inv = (lrow[r] > 0.f) ? (1.0f / lrow[r]) : 0.f;
    const size_t base = ((size_t)b * LQ + li) * DM + (size_t)h * DH;
    for (int nc = 0; nc < 8; ++nc)
      attn[base + nc * 16 + ll] = f2bf(oacc[nc][r] * inv);
  }
}

// =====================================================================
// Kernel 3: output projection. A (attn, bf16, row-major) is staged with
// gfx1250 GLOBAL_LOAD_ASYNC_TO_LDS_B128 (ASYNCcnt-tracked, LDS double-
// buffered); B (Wo fp32) is register-prefetched and converted to bf16.
// grid: (32 N-tiles, 4 M-tiles), 256 thr.
// =====================================================================
__global__ __launch_bounds__(256)
void out_kernel(const unsigned short* __restrict__ attn, const float* __restrict__ Wo,
                float* __restrict__ out)
{
  const int nt = blockIdx.x, mt = blockIdx.y;
  const int m0 = mt * 128, n0 = nt * 128;

  __shared__ unsigned short As[2][128][40];  // async-filled, double buffered
  __shared__ unsigned short Bs[128][40];

  const int tid = threadIdx.x, lane = tid & 31, wv = tid >> 5;
  const int hh = lane >> 4, ll = lane & 15;

  // A async staging: 512 x 16B chunks per tile, 2 per thread
  const int cid0 = tid * 2;
  const int ar0 = (cid0 + 0) >> 2, ac0 = ((cid0 + 0) & 3) * 8;
  const int ar1 = (cid0 + 1) >> 2, ac1 = ((cid0 + 1) & 3) * 8;

  const int b_c4 = (tid & 31) * 4;
  const int b_kr = tid >> 5;

  v8f acc[8] = {};
  float4 rB[4];

  // issue async copy of A tile t into As[t&1]
  auto issueA = [&](int k0, int buf) {
    const unsigned short* g0 = &attn[(size_t)(m0 + ar0) * DM + k0 + ac0];
    const unsigned short* g1 = &attn[(size_t)(m0 + ar1) * DM + k0 + ac1];
    unsigned l0 = (unsigned)(size_t)&As[buf][ar0][ac0];
    unsigned l1 = (unsigned)(size_t)&As[buf][ar1][ac1];
    asm volatile("global_load_async_to_lds_b128 %0, %1, off"
                 :: "v"(l0), "v"(g0) : "memory");
    asm volatile("global_load_async_to_lds_b128 %0, %1, off"
                 :: "v"(l1), "v"(g1) : "memory");
  };

  // prologue: async A(0), prefetch B(0)
  issueA(0, 0);
  for (int p = 0; p < 4; ++p)
    rB[p] = *reinterpret_cast<const float4*>(&Wo[(size_t)(b_kr + 8 * p) * DM + n0 + b_c4]);

  for (int k0 = 0; k0 < DM; k0 += 32) {
    const int t   = k0 >> 5;
    const bool more = (k0 + 32 < DM);
    __syncthreads();   // prev compute done: Bs and As[(t+1)&1] free
    // store B(t) regs -> LDS (transposed, fp32 -> bf16)
    for (int p = 0; p < 4; ++p) {
      const int k = b_kr + 8 * p;
      Bs[b_c4 + 0][k] = f2bft(rB[p].x); Bs[b_c4 + 1][k] = f2bft(rB[p].y);
      Bs[b_c4 + 2][k] = f2bft(rB[p].z); Bs[b_c4 + 3][k] = f2bft(rB[p].w);
    }
    if (more) {
      issueA(k0 + 32, (t + 1) & 1);   // overlaps this tile's compute
      for (int p = 0; p < 4; ++p)
        rB[p] = *reinterpret_cast<const float4*>(&Wo[(size_t)(k0 + 32 + b_kr + 8 * p) * DM + n0 + b_c4]);
      // allow the 2 in-flight instrs of tile t+1; async completes in order,
      // so <=2 outstanding guarantees tile t has landed in LDS
      asm volatile("s_wait_asynccnt 2" ::: "memory");
    } else {
      asm volatile("s_wait_asynccnt 0" ::: "memory");
    }
    __syncthreads();   // staging visible to all waves

    Frag af;
    const int row = wv * 16 + ll;
    const int koff = hh * 8;
    af.u[0] = *reinterpret_cast<const uint4*>(&As[t & 1][row][koff]);
    af.u[1] = *reinterpret_cast<const uint4*>(&As[t & 1][row][koff + 16]);
    for (int nc = 0; nc < 8; ++nc) {
      Frag bf_;
      const int n = nc * 16 + ll;
      const int kb2 = hh * 8;
      bf_.u[0] = *reinterpret_cast<const uint4*>(&Bs[n][kb2]);
      bf_.u[1] = *reinterpret_cast<const uint4*>(&Bs[n][kb2 + 16]);
      acc[nc] = WMMA_BF16(af.v, bf_.v, acc[nc]);
    }
  }

  const int rowg = m0 + wv * 16 + hh * 8;
  for (int r = 0; r < 8; ++r)
    for (int nc = 0; nc < 8; ++nc)
      out[(size_t)(rowg + r) * DM + n0 + nc * 16 + ll] = acc[nc][r];
}

// =====================================================================
extern "C" void kernel_launch(void* const* d_in, const int* in_sizes, int n_in,
                              void* d_out, int out_size, void* d_ws, size_t ws_size,
                              hipStream_t stream)
{
  (void)in_sizes; (void)n_in; (void)out_size; (void)ws_size;
  const float* x       = (const float*)d_in[0];
  const float* cache_k = (const float*)d_in[1];
  const float* cache_v = (const float*)d_in[2];
  const float* Wq      = (const float*)d_in[3];
  const float* Wk      = (const float*)d_in[4];
  const float* Wv      = (const float*)d_in[5];
  const float* Wo      = (const float*)d_in[6];
  const float* qw      = (const float*)d_in[7];
  const float* kw      = (const float*)d_in[8];

  const size_t HE = (size_t)BD * NH * LQ * DH;       // per-buffer bf16 elems
  unsigned short* qb   = (unsigned short*)d_ws;
  unsigned short* kb   = qb + HE;
  unsigned short* vb   = kb + HE;
  unsigned short* attn = vb + HE;                    // [B*L, D] bf16

  qkv_kernel<<<dim3(96, 4), 256, 0, stream>>>(x, Wq, Wk, Wv, qw, kw, qb, kb, vb);
  attn_kernel<<<dim3(128), 256, 0, stream>>>(cache_k, cache_v, qb, kb, vb, attn);
  out_kernel<<<dim3(32, 4), 256, 0, stream>>>(attn, Wo, (float*)d_out);
}
